// NeuralMemory_85057532330661
// MI455X (gfx1250) — compile-verified, matrix operand order checked by reading, not run on hardware
//
#include <hip/hip_runtime.h>

// ---------------------------------------------------------------------------
// CDNA5 (gfx1250, wave32) implementation of the NeuralMemory reference.
// All matmuls run on v_wmma_f32_16x16x32_bf16 (fp32->bf16 at LDS staging,
// fp32 accumulation). Attention is flash-style with WMMA for QK^T and P@V.
// Fragments are fetched as 2x ds_load_b128 per the ISA 16-bit 16x32 layout:
// lanes 0-15 hold K[0..7]+[16..23], lanes 16-31 hold K[8..15]+[24..31].
// ---------------------------------------------------------------------------

typedef __bf16 bf16_t;
typedef __attribute__((ext_vector_type(16))) __bf16 v16bf;
typedef __attribute__((ext_vector_type(8)))  __bf16 v8bf;
typedef __attribute__((ext_vector_type(2)))  __bf16 v2bf;
typedef __attribute__((ext_vector_type(8)))  float  v8f;

#define WMMA_BF16(A, B, C) \
  __builtin_amdgcn_wmma_f32_16x16x32_bf16(false, (A), false, (B), (short)0, (C), false, false)

// ---- constants from the reference -----------------------------------------
#define DD      512
#define SEQ     2048
#define BATCH   4
#define TOK     (BATCH * SEQ)          // 8192
#define TD      ((size_t)TOK * DD)     // 4,194,304 elements
#define NCHUNK  16
#define CHL     128
#define HEADS   8
#define HD      64
#define WIN     256
#define LRATE   0.01f
#define WDEC    0.01f
#define MOM     0.9f
#define MAXALR  0.1f
#define EPSV    1e-8f

__device__ __forceinline__ float sigmoidf_(float x) { return 1.0f / (1.0f + __expf(-x)); }
__device__ __forceinline__ float siluf_(float x)    { return x * sigmoidf_(x); }
__device__ __forceinline__ float dsiluf_(float x)   { float s = sigmoidf_(x); return s * (1.0f + x * (1.0f - s)); }

__device__ __forceinline__ v2bf pack2(float a, float b) {
  v2bf t; t[0] = (bf16_t)a; t[1] = (bf16_t)b; return t;
}

// Fragment load: slice i = lane&15 along the outer LDS dim, K contiguous
// inner. Two 16-byte LDS loads per lane (ld must be a multiple of 8).
__device__ __forceinline__ v16bf frag16(const bf16_t* p, int ld) {
  int lane = threadIdx.x & 31;
  int i = lane & 15, hf = lane >> 4;
  const bf16_t* base = p + i * ld + hf * 8;
  v8bf lo = *(const v8bf*)(base);
  v8bf hi = *(const v8bf*)(base + 16);
  return __builtin_shufflevector(lo, hi, 0, 1, 2, 3, 4, 5, 6, 7,
                                 8, 9, 10, 11, 12, 13, 14, 15);
}

// ---------------------------------------------------------------------------
// Generic WMMA GEMM: C = epi(A[MxK] @ op(B)); macro tile 128x64, 8 waves,
// each wave 32x32 (2x2 WMMA tiles), K-step 32.
// LDS: A row-major [m][k], B column-major [n][k], pitch 40 (80 B, 16B-mult).
// ---------------------------------------------------------------------------
#define BM 128
#define BN 64
#define BK 32
#define LDK 40

enum { EPI_NONE = 0, EPI_BIAS = 1, EPI_SILU_RES = 2, EPI_RES = 3 };

template <int EPI, bool TRB, bool STORE_PRE>
__global__ __launch_bounds__(256) void gemm_kernel(
    const float* __restrict__ A, const float* __restrict__ Bm,
    const float* __restrict__ bias, const float* __restrict__ res,
    float* __restrict__ out, float* __restrict__ pre,
    int M, int N, int K) {
  __shared__ alignas(16) bf16_t As[BM][LDK];   // [m][k]
  __shared__ alignas(16) bf16_t Bs[BN][LDK];   // [n][k] (column-major B)

  const int tid = threadIdx.x;
  const int wave = tid >> 5;
  const int wm = (wave >> 1) * 32;
  const int wn = (wave & 1) * 32;
  const int bm = blockIdx.x * BM;
  const int bn = blockIdx.y * BN;

  v8f acc[2][2];
#pragma unroll
  for (int mi = 0; mi < 2; ++mi)
#pragma unroll
    for (int ni = 0; ni < 2; ++ni)
#pragma unroll
      for (int e = 0; e < 8; ++e) acc[mi][ni][e] = 0.0f;

  for (int k0 = 0; k0 < K; k0 += BK) {
    // stage A: 128x32, float2 loads (coalesced on k), packed v2bf stores
#pragma unroll
    for (int i = 0; i < 8; ++i) {
      int ep = tid + i * 256;           // 2048 k-pairs
      int r = ep >> 4, pc = ep & 15;
      float2 f = *(const float2*)&A[(size_t)(bm + r) * K + k0 + 2 * pc];
      *(v2bf*)&As[r][2 * pc] = pack2(f.x, f.y);
    }
    // stage B column-major
#pragma unroll
    for (int i = 0; i < 4; ++i) {
      int ep = tid + i * 256;           // 1024 pairs
      if (!TRB) {
        int c = ep & 63, rp = ep >> 6;  // pair along k, strided global reads
        float f0 = Bm[(size_t)(k0 + 2 * rp) * N + bn + c];
        float f1 = Bm[(size_t)(k0 + 2 * rp + 1) * N + bn + c];
        *(v2bf*)&Bs[c][2 * rp] = pack2(f0, f1);
      } else {                          // W[n][k]: natural, fully coalesced
        int n = ep >> 4, pc = ep & 15;
        float2 f = *(const float2*)&Bm[(size_t)(bn + n) * K + k0 + 2 * pc];
        *(v2bf*)&Bs[n][2 * pc] = pack2(f.x, f.y);
      }
    }
    if (k0 + BK < K)  // prefetch next A K-tile (global_prefetch_b8)
      __builtin_prefetch(&A[(size_t)(bm + (tid >> 1)) * K + k0 + BK], 0, 1);
    __syncthreads();

    v16bf a0 = frag16(&As[wm][0], LDK);
    v16bf a1 = frag16(&As[wm + 16][0], LDK);
    v16bf b0 = frag16(&Bs[wn][0], LDK);
    v16bf b1 = frag16(&Bs[wn + 16][0], LDK);
    acc[0][0] = WMMA_BF16(a0, b0, acc[0][0]);
    acc[0][1] = WMMA_BF16(a0, b1, acc[0][1]);
    acc[1][0] = WMMA_BF16(a1, b0, acc[1][0]);
    acc[1][1] = WMMA_BF16(a1, b1, acc[1][1]);
    __syncthreads();
  }

  // epilogue: C layout -> row = r + (lane>>4)*8, col = lane&15
  const int lane = tid & 31;
  const int hf = lane >> 4;
  const int cl = lane & 15;
#pragma unroll
  for (int mi = 0; mi < 2; ++mi)
#pragma unroll
    for (int ni = 0; ni < 2; ++ni) {
      int cgl = bn + wn + ni * 16 + cl;
#pragma unroll
      for (int r = 0; r < 8; ++r) {
        int rgl = bm + wm + mi * 16 + hf * 8 + r;
        size_t o = (size_t)rgl * N + cgl;
        float v = acc[mi][ni][r];
        if (EPI == EPI_BIAS) v += bias[cgl];
        if (STORE_PRE) pre[o] = v;
        if (EPI == EPI_SILU_RES) v = res[o] + siluf_(v);
        if (EPI == EPI_RES) v = v + res[o];
        out[o] = v;
      }
    }
}

// ---------------------------------------------------------------------------
// Per-chunk weight grad: U[c] = Act_c^T @ Grad_c (512x512, K = 4 batches x
// 128 tokens of chunk c). grid = (4, 8, 16 chunks).
// LDS: As[m][t] (A^T row-major), Bs[n][t] (column-major Grad).
// ---------------------------------------------------------------------------
__global__ __launch_bounds__(256) void wgrad_kernel(
    const float* __restrict__ Act, const float* __restrict__ Grad,
    float* __restrict__ U) {
  __shared__ alignas(16) bf16_t As[BM][LDK];
  __shared__ alignas(16) bf16_t Bs[BN][LDK];

  const int tid = threadIdx.x;
  const int wave = tid >> 5;
  const int wm = (wave >> 1) * 32;
  const int wn = (wave & 1) * 32;
  const int bm = blockIdx.x * BM;
  const int bn = blockIdx.y * BN;
  const int c = blockIdx.z;
  float* __restrict__ out = U + (size_t)c * DD * DD;

  v8f acc[2][2];
#pragma unroll
  for (int mi = 0; mi < 2; ++mi)
#pragma unroll
    for (int ni = 0; ni < 2; ++ni)
#pragma unroll
      for (int e = 0; e < 8; ++e) acc[mi][ni][e] = 0.0f;

  for (int b = 0; b < BATCH; ++b) {
    const int tbase = b * SEQ + c * CHL;
    for (int kb = 0; kb < CHL; kb += BK) {
      // As[m][t] = Act[t][bm+m]; token pairs, coalesced on m
#pragma unroll
      for (int i = 0; i < 8; ++i) {
        int ep = tid + i * 256;          // 2048 pairs (128 m x 16 t-pairs)
        int tp = ep >> 7, m = ep & 127;
        float f0 = Act[(size_t)(tbase + kb + 2 * tp) * DD + bm + m];
        float f1 = Act[(size_t)(tbase + kb + 2 * tp + 1) * DD + bm + m];
        *(v2bf*)&As[m][2 * tp] = pack2(f0, f1);
      }
      // Bs[n][t] = Grad[t][bn+n]
#pragma unroll
      for (int i = 0; i < 4; ++i) {
        int ep = tid + i * 256;          // 1024 pairs (64 n x 16 t-pairs)
        int tp = ep >> 6, n = ep & 63;
        float f0 = Grad[(size_t)(tbase + kb + 2 * tp) * DD + bn + n];
        float f1 = Grad[(size_t)(tbase + kb + 2 * tp + 1) * DD + bn + n];
        *(v2bf*)&Bs[n][2 * tp] = pack2(f0, f1);
      }
      __syncthreads();
      v16bf a0 = frag16(&As[wm][0], LDK);
      v16bf a1 = frag16(&As[wm + 16][0], LDK);
      v16bf b0 = frag16(&Bs[wn][0], LDK);
      v16bf b1 = frag16(&Bs[wn + 16][0], LDK);
      acc[0][0] = WMMA_BF16(a0, b0, acc[0][0]);
      acc[0][1] = WMMA_BF16(a0, b1, acc[0][1]);
      acc[1][0] = WMMA_BF16(a1, b0, acc[1][0]);
      acc[1][1] = WMMA_BF16(a1, b1, acc[1][1]);
      __syncthreads();
    }
  }
  const int lane = tid & 31;
  const int hf = lane >> 4, cl = lane & 15;
#pragma unroll
  for (int mi = 0; mi < 2; ++mi)
#pragma unroll
    for (int ni = 0; ni < 2; ++ni) {
      int cgl = bn + wn + ni * 16 + cl;
#pragma unroll
      for (int r = 0; r < 8; ++r) {
        int rgl = bm + wm + mi * 16 + hf * 8 + r;
        out[(size_t)rgl * DD + cgl] = acc[mi][ni][r];
      }
    }
}

// ---------------------------------------------------------------------------
// Flash-style causal sliding-window attention (window 256), WMMA for both
// QK^T and P@V. One block = (64 queries, one head, one batch), 4 waves.
// LDS: Qs[q][d] (A), Ks[key][d] (col-major B of QK^T, no transpose needed),
// Vs[d][key] (col-major B of P@V), Ps wave-private P tiles.
// ---------------------------------------------------------------------------
__global__ __launch_bounds__(128) void attn_kernel(
    const float* __restrict__ Q, const float* __restrict__ Kp,
    const float* __restrict__ Vp, float* __restrict__ O) {
  __shared__ alignas(16) bf16_t Qs[64][72];
  __shared__ alignas(16) bf16_t Ks[32][72];
  __shared__ alignas(16) bf16_t Vs[64][40];
  __shared__ alignas(16) bf16_t Ps[4][16][40];

  const int tid = threadIdx.x;
  const int wave = tid >> 5;
  const int lane = tid & 31;
  const int hf = lane >> 4;
  const int ln = lane & 15;
  const int qbase = blockIdx.x * 64;
  const int h = blockIdx.y;
  const int b = blockIdx.z;
  const size_t rowoff = (size_t)b * SEQ;
  const int colh = h * HD;

  // stage 64x64 Q tile (d-pairs, coalesced)
#pragma unroll
  for (int i = 0; i < 16; ++i) {
    int ep = tid + i * 128;              // 2048 pairs
    int qr = ep >> 5, dp = ep & 31;
    float2 f = *(const float2*)&Q[(rowoff + qbase + qr) * DD + colh + 2 * dp];
    *(v2bf*)&Qs[qr][2 * dp] = pack2(f.x, f.y);
  }
  __syncthreads();
  v16bf aq0 = frag16(&Qs[wave * 16][0], 72);
  v16bf aq1 = frag16(&Qs[wave * 16][32], 72);

  float mrun[8], lrun[8];
  v8f oacc[4];
#pragma unroll
  for (int r = 0; r < 8; ++r) { mrun[r] = -1e30f; lrun[r] = 0.0f; }
#pragma unroll
  for (int t = 0; t < 4; ++t)
#pragma unroll
    for (int e = 0; e < 8; ++e) oacc[t][e] = 0.0f;

  int ks0 = qbase - (WIN - 1);
  if (ks0 < 0) ks0 = 0;
  ks0 &= ~31;

  for (int kb = ks0; kb < qbase + 64; kb += 32) {
    // Ks[key][d]: natural layout, d-pairs coalesced
#pragma unroll
    for (int i = 0; i < 8; ++i) {
      int ep = tid + i * 128;            // 1024 pairs (32 keys x 32 d-pairs)
      int kk = ep >> 5, dp = ep & 31;
      float2 f = *(const float2*)&Kp[(rowoff + kb + kk) * DD + colh + 2 * dp];
      *(v2bf*)&Ks[kk][2 * dp] = pack2(f.x, f.y);
    }
    // Vs[d][key]: transposed, key-pairs, coalesced on d
#pragma unroll
    for (int i = 0; i < 8; ++i) {
      int ep = tid + i * 128;            // 1024 pairs (64 d x 16 key-pairs)
      int kt = ep >> 6, d = ep & 63;
      float f0 = Vp[(rowoff + kb + 2 * kt) * DD + colh + d];
      float f1 = Vp[(rowoff + kb + 2 * kt + 1) * DD + colh + d];
      *(v2bf*)&Vs[d][2 * kt] = pack2(f0, f1);
    }
    __syncthreads();

    v8f s0, s1;
#pragma unroll
    for (int e = 0; e < 8; ++e) { s0[e] = 0.0f; s1[e] = 0.0f; }
    s0 = WMMA_BF16(aq0, frag16(&Ks[0][0], 72), s0);
    s0 = WMMA_BF16(aq1, frag16(&Ks[0][32], 72), s0);
    s1 = WMMA_BF16(aq0, frag16(&Ks[16][0], 72), s1);
    s1 = WMMA_BF16(aq1, frag16(&Ks[16][32], 72), s1);

#pragma unroll
    for (int r = 0; r < 8; ++r) {
      int q = qbase + wave * 16 + hf * 8 + r;
      int key0 = kb + ln, key1 = kb + 16 + ln;
      bool ok0 = (key0 <= q) && (q - key0 < WIN);
      bool ok1 = (key1 <= q) && (q - key1 < WIN);
      float v0 = ok0 ? s0[r] * 0.125f : -1e30f;  // 1/sqrt(64)
      float v1 = ok1 ? s1[r] * 0.125f : -1e30f;
      float sm = fmaxf(v0, v1);
#pragma unroll
      for (int off = 1; off < 16; off <<= 1) sm = fmaxf(sm, __shfl_xor(sm, off, 32));
      float mnew = fmaxf(mrun[r], sm);
      float sc = __expf(mrun[r] - mnew);
      float p0 = ok0 ? __expf(v0 - mnew) : 0.0f;
      float p1 = ok1 ? __expf(v1 - mnew) : 0.0f;
      float rs = p0 + p1;
#pragma unroll
      for (int off = 1; off < 16; off <<= 1) rs += __shfl_xor(rs, off, 32);
      lrun[r] = lrun[r] * sc + rs;
      mrun[r] = mnew;
#pragma unroll
      for (int t = 0; t < 4; ++t) oacc[t][r] *= sc;
      Ps[wave][hf * 8 + r][ln] = (bf16_t)p0;
      Ps[wave][hf * 8 + r][16 + ln] = (bf16_t)p1;
    }

    v16bf ap = frag16(&Ps[wave][0][0], 40);
#pragma unroll
    for (int t = 0; t < 4; ++t)
      oacc[t] = WMMA_BF16(ap, frag16(&Vs[t * 16][0], 40), oacc[t]);
    __syncthreads();
  }

#pragma unroll
  for (int r = 0; r < 8; ++r) {
    float inv = 1.0f / lrun[r];
    int q = qbase + wave * 16 + hf * 8 + r;
#pragma unroll
    for (int t = 0; t < 4; ++t)
      O[(rowoff + q) * DD + colh + t * 16 + ln] = oacc[t][r] * inv;
  }
}

// ---------------------------------------------------------------------------
// Elementwise / reduction helpers
// ---------------------------------------------------------------------------
__global__ void concat_kernel(const float* __restrict__ x,
                              const float* __restrict__ meta,
                              float* __restrict__ xm) {
  size_t i = (size_t)blockIdx.x * 256 + threadIdx.x;
  int d = (int)(i & 511);
  size_t tg = i >> 9;
  int b = (int)(tg >> 11), s = (int)(tg & 2047);
  xm[i] = (s < 64) ? meta[(size_t)s * DD + d]
                   : x[((size_t)b * 1984 + (s - 64)) * DD + d];
}

__global__ void alr_kernel(const float* __restrict__ xm,
                           const float* __restrict__ Wlr,
                           const float* __restrict__ blr,
                           float* __restrict__ alr) {
  int tok = blockIdx.x * 8 + (threadIdx.x >> 5);
  int lane = threadIdx.x & 31;
  const float* row = xm + (size_t)tok * DD;
  float s = 0.0f;
#pragma unroll
  for (int i = 0; i < 16; ++i) s += row[lane * 16 + i] * Wlr[lane * 16 + i];
#pragma unroll
  for (int off = 16; off; off >>= 1) s += __shfl_xor(s, off, 32);
  if (lane == 0) alr[tok] = MAXALR * sigmoidf_(s + blr[0]);
}

__global__ void d2_kernel(const float* __restrict__ z2,
                          const float* __restrict__ val,
                          const float* __restrict__ h1,
                          const float* __restrict__ alr,
                          float* __restrict__ d2, float* __restrict__ dh1) {
  size_t i = (size_t)blockIdx.x * 256 + threadIdx.x;
  float w = alr[i >> 9];
  float g = (2.0f / (float)DD) * w * (z2[i] - val[i]);
  d2[i] = g;
  dh1[i] = g * dsiluf_(h1[i]);
}

__global__ void dh0_kernel(float* __restrict__ dz1, const float* __restrict__ h0) {
  size_t i = (size_t)blockIdx.x * 256 + threadIdx.x;
  dz1[i] = dz1[i] * dsiluf_(h0[i]);
}

// wcoef[c] = (1 - coef[c]*(16-c))/16, coef[c] = mean(alr over chunk)*MOM^(c+1)
__global__ void wcoef_kernel(const float* __restrict__ alr, float* __restrict__ wcoef) {
  int c = blockIdx.x;
  int l = threadIdx.x;  // 128 threads
  float s = 0.0f;
#pragma unroll
  for (int b = 0; b < BATCH; ++b) s += alr[b * SEQ + c * CHL + l];
  __shared__ float red[4];
#pragma unroll
  for (int off = 16; off; off >>= 1) s += __shfl_xor(s, off, 32);
  if ((l & 31) == 0) red[l >> 5] = s;
  __syncthreads();
  if (l == 0) {
    float tot = red[0] + red[1] + red[2] + red[3];
    float coef = (tot / (float)(BATCH * CHL)) * __powf(MOM, (float)(c + 1));
    wcoef[c] = (1.0f - coef * (float)(NCHUNK - c)) / (float)NCHUNK;
  }
}

__global__ void update_kernel(const float* __restrict__ u,
                              const float* __restrict__ wcoef,
                              const float* __restrict__ w0,
                              float* __restrict__ nw) {
  size_t i = (size_t)blockIdx.x * 256 + threadIdx.x;  // 2*512*512
  size_t layer = i >> 18;
  size_t e = i & ((1u << 18) - 1);
  float g = 0.0f;
#pragma unroll
  for (int c = 0; c < NCHUNK; ++c)
    g += u[(layer * NCHUNK + c) * (size_t)(DD * DD) + e] * wcoef[c];
  float w = w0[i];
  float upd = LRATE * g / (fabsf(g) + EPSV) + LRATE * WDEC * w;
  nw[i] = w - upd;
}

__global__ void copyout_kernel(const float* __restrict__ buf, float* __restrict__ out) {
  size_t i = (size_t)blockIdx.x * 256 + threadIdx.x;  // 4*1984*512
  int d = (int)(i & 511);
  size_t t = i >> 9;
  int b = (int)(t / 1984), s = (int)(t % 1984);
  out[i] = buf[((size_t)b * SEQ + 64 + s) * DD + d];
}

// ---------------------------------------------------------------------------
// Orchestration
// ---------------------------------------------------------------------------
extern "C" void kernel_launch(void* const* d_in, const int* in_sizes, int n_in,
                              void* d_out, int out_size, void* d_ws, size_t ws_size,
                              hipStream_t stream) {
  const float* x      = (const float*)d_in[0];
  const float* meta   = (const float*)d_in[1];
  const float* lmm_w  = (const float*)d_in[2];
  const float* Wq     = (const float*)d_in[3];
  const float* bq     = (const float*)d_in[4];
  const float* Wk     = (const float*)d_in[5];
  const float* bk     = (const float*)d_in[6];
  const float* Wv     = (const float*)d_in[7];
  const float* bv     = (const float*)d_in[8];
  const float* Wlr    = (const float*)d_in[9];
  const float* blr    = (const float*)d_in[10];
  const float* sWq    = (const float*)d_in[11];
  const float* sWk    = (const float*)d_in[12];
  const float* sWv    = (const float*)d_in[13];
  const float* sWo    = (const float*)d_in[14];

  float* ws = (float*)d_ws;
  float* xm      = ws + 0 * TD;
  float* keys    = ws + 1 * TD;
  float* values  = ws + 2 * TD;
  float* queries = ws + 3 * TD;
  float* h0      = ws + 4 * TD;
  float* z1      = ws + 5 * TD;
  float* h1      = ws + 6 * TD;
  float* z2      = ws + 7 * TD;
  float* d2b     = ws + 8 * TD;
  float* dh1     = ws + 9 * TD;
  float* dz1     = ws + 10 * TD;
  float* u       = ws + 11 * TD;          // 2*16*512*512 = 2*TD floats
  float* new_ws  = ws + 13 * TD;          // 2*512*512
  float* alr     = new_ws + 2 * DD * DD;  // 8192
  float* wcoef   = alr + TOK;             // 16
  // buffer reuse (lifetimes disjoint):
  float* r1      = h0;
  float* retr    = h1;
  float* qp      = z2;
  float* kp      = d2b;
  float* vp      = dh1;
  float* attnout = dz1;
  float* outbuf  = xm;

  const size_t W2 = (size_t)DD * DD;
  dim3 gG(TOK / BM, DD / BN);         // (64, 8) big GEMMs
  dim3 gW(DD / BM, DD / BN, NCHUNK);  // (4, 8, 16) weight grads
  dim3 gA(SEQ / 64, HEADS, BATCH);    // (32, 8, 4) attention

  // 1. build xm, projections, adaptive LR
  concat_kernel<<<TD / 256, 256, 0, stream>>>(x, meta, xm);
  gemm_kernel<EPI_BIAS, false, false><<<gG, 256, 0, stream>>>(xm, Wk, bk, nullptr, keys, nullptr, TOK, DD, DD);
  gemm_kernel<EPI_BIAS, false, false><<<gG, 256, 0, stream>>>(xm, Wv, bv, nullptr, values, nullptr, TOK, DD, DD);
  gemm_kernel<EPI_BIAS, false, false><<<gG, 256, 0, stream>>>(xm, Wq, bq, nullptr, queries, nullptr, TOK, DD, DD);
  alr_kernel<<<TOK / 8, 256, 0, stream>>>(xm, Wlr, blr, alr);

  // 2. chunk MLP forward (save pre-activations)
  gemm_kernel<EPI_SILU_RES, false, true><<<gG, 256, 0, stream>>>(keys, lmm_w, nullptr, keys, z1, h0, TOK, DD, DD);
  gemm_kernel<EPI_SILU_RES, false, true><<<gG, 256, 0, stream>>>(z1, lmm_w + W2, nullptr, z1, z2, h1, TOK, DD, DD);

  // 3. backward -> per-chunk weight grads u
  d2_kernel<<<TD / 256, 256, 0, stream>>>(z2, values, h1, alr, d2b, dh1);
  gemm_kernel<EPI_RES, true, false><<<gG, 256, 0, stream>>>(dh1, lmm_w + W2, nullptr, d2b, dz1, nullptr, TOK, DD, DD);
  dh0_kernel<<<TD / 256, 256, 0, stream>>>(dz1, h0);
  wgrad_kernel<<<gW, 256, 0, stream>>>(keys, dz1, u);                 // layer 0
  wgrad_kernel<<<gW, 256, 0, stream>>>(z1, dh1, u + NCHUNK * W2);     // layer 1

  // 4. folded cumsum + AdamW-style step
  wcoef_kernel<<<NCHUNK, 128, 0, stream>>>(alr, wcoef);
  update_kernel<<<(2 * W2) / 256, 256, 0, stream>>>(u, wcoef, lmm_w, new_ws);

  // 5. retrieval with updated memory
  gemm_kernel<EPI_SILU_RES, false, false><<<gG, 256, 0, stream>>>(queries, new_ws, nullptr, queries, r1, nullptr, TOK, DD, DD);
  gemm_kernel<EPI_SILU_RES, false, false><<<gG, 256, 0, stream>>>(r1, new_ws + W2, nullptr, r1, retr, nullptr, TOK, DD, DD);

  // 6. sliding-window attention
  gemm_kernel<EPI_NONE, false, false><<<gG, 256, 0, stream>>>(retr, sWq, nullptr, nullptr, qp, nullptr, TOK, DD, DD);
  gemm_kernel<EPI_NONE, false, false><<<gG, 256, 0, stream>>>(retr, sWk, nullptr, nullptr, kp, nullptr, TOK, DD, DD);
  gemm_kernel<EPI_NONE, false, false><<<gG, 256, 0, stream>>>(retr, sWv, nullptr, nullptr, vp, nullptr, TOK, DD, DD);
  attn_kernel<<<gA, 128, 0, stream>>>(qp, kp, vp, attnout);
  gemm_kernel<EPI_NONE, false, false><<<gG, 256, 0, stream>>>(attnout, sWo, nullptr, nullptr, outbuf, nullptr, TOK, DD, DD);

  // 7. strip meta tokens into d_out
  copyout_kernel<<<((size_t)BATCH * 1984 * DD) / 256, 256, 0, stream>>>(outbuf, (float*)d_out);
}
